// FmtsSelfAttention_2095944040637
// MI455X (gfx1250) — matrix operand order repaired and checked by reading, not compiled, not run on hardware
//
#include <hip/hip_runtime.h>
#include <hip/hip_bf16.h>

typedef __bf16 bf16;
typedef __attribute__((ext_vector_type(16))) __bf16 v16bf;
typedef __attribute__((ext_vector_type(8)))  float  v8f;
typedef __attribute__((ext_vector_type(4)))  int    v4i;

// Async global->LDS builtin takes v4i pointers (per hipcc diagnostic):
typedef __attribute__((address_space(1))) v4i gv4i;
typedef __attribute__((address_space(3))) v4i lv4i;

#if defined(__has_builtin)
#if __has_builtin(__builtin_amdgcn_global_load_async_to_lds_b128) && \
    __has_builtin(__builtin_amdgcn_s_wait_asynccnt)
#define HAVE_ASYNC_LDS 1
#endif
#endif

static constexpr int B_ = 8, T_ = 1024, E_ = 256, H_ = 8, HE_ = 2048;

// ---- WMMA fragment builders (CDNA5 wave32 layouts, CDNA5 ISA) ----
__device__ __forceinline__ v16bf frag_a(const bf16* p, int ld) {
  int lane = threadIdx.x & 31;
  int m = lane & 15, hlf = lane >> 4;
  const bf16* row = p + m * ld;
  v16bf a;
#pragma unroll
  for (int v = 0; v < 8; ++v) {
    int k = hlf * 16 + 2 * v;
    a[2 * v]     = row[k];
    a[2 * v + 1] = row[k + 1];
  }
  return a;
}

// B fragment from transposed source: element(k,n) = p[n*ld + k]  (for Q.K^T).
__device__ __forceinline__ v16bf frag_b_t(const bf16* p, int ld) {
  int lane = threadIdx.x & 31;
  int n = lane & 15, hlf = lane >> 4;
  const bf16* row = p + (size_t)n * ld + hlf * 16;
  v16bf b;
#pragma unroll
  for (int v = 0; v < 8; ++v) {
    b[2 * v]     = row[2 * v];
    b[2 * v + 1] = row[2 * v + 1];
  }
  return b;
}

// Swizzled B tile (K=32 x N=16): lane (k/16)*16+n holds its 16 K-values
// contiguously, stride 24 elems (48B, 16B aligned) -> 2x ds_load_b128.
__device__ __forceinline__ v16bf frag_b_swz(const bf16* tile) {
  int lane = threadIdx.x & 31;
  const bf16* p = tile + lane * 24;
  v16bf b;
#pragma unroll
  for (int v = 0; v < 8; ++v) {
    b[2 * v]     = p[2 * v];
    b[2 * v + 1] = p[2 * v + 1];
  }
  return b;
}

#define WMMA_BF16(a, b, c) \
  __builtin_amdgcn_wmma_f32_16x16x32_bf16(false, (a), false, (b), (short)0, (c), false, false)

// ============================================================================
// Kernel 1: QKV projections. out[bh][t][o] = sum_d x[b][t][d] * W[h*E+o][d]
// 64x64 tile, 8 waves x two 16x16 WMMA accumulators. Epilogue staged through
// LDS for coalesced b128 stores; V written in fragment order for flash.
// ============================================================================
__global__ void qkv_kernel(const float* __restrict__ x,
                           const float* __restrict__ Wq,
                           const float* __restrict__ Wk,
                           const float* __restrict__ Wv,
                           bf16* __restrict__ Qb, bf16* __restrict__ Kb,
                           bf16* __restrict__ Vb) {
  int idx = blockIdx.x;
  int nb = idx & 3;  idx >>= 2;   // E/64
  int tb = idx & 15; idx >>= 4;   // T/64
  int bh = idx & 63; idx >>= 6;   // B*H
  int which = idx;                // 0=Q,1=K,2=V
  int b = bh >> 3, h = bh & 7;
  const float* W  = (which == 0) ? Wq : (which == 1) ? Wk : Wv;
  bf16*        Out = (which == 0) ? Qb : Kb;

  // Manual LDS layout so the epilogue f32 staging overlays As/Bs.
  __shared__ alignas(16) unsigned char smq[11264];
  bf16 (*As)[40]     = (bf16(*)[40])smq;                 // 5120 B
  bf16 (*Bs)[32][24] = (bf16(*)[32][24])(smq + 5120);    // 6144 B
  float (*Fst)[68]   = (float(*)[68])smq;                // epilogue: 8704 B

  int tid = threadIdx.x, lane = tid & 31, wave = tid >> 5;
  int wr = wave >> 1, wc = wave & 1;
  int tbase = tb * 64, obase = nb * 64;
  int nl = lane & 15, hlf = lane >> 4;

  const float* xb = x + ((size_t)b * T_ + tbase) * E_;
  const float* wb = W + ((size_t)h * E_ + obase) * E_;

  v8f acc0 = {}, acc1 = {};
  for (int d0 = 0; d0 < E_; d0 += 32) {
    __syncthreads();
#pragma unroll
    for (int i = 0; i < 2; ++i) {            // A tile 64x32 fp32 -> bf16
      int lin = i * 256 + tid;
      int r = lin >> 3, c4 = lin & 7;
      float4 f = *(const float4*)(xb + (size_t)r * E_ + d0 + c4 * 4);
      alignas(8) bf16 t[4] = {(bf16)f.x, (bf16)f.y, (bf16)f.z, (bf16)f.w};
      *(uint2*)&As[r][c4 * 4] = *(const uint2*)t;
    }
#pragma unroll
    for (int i = 0; i < 2; ++i) {            // B tile (W^T) 32x64, swizzled
      int lin = i * 256 + tid;
      int nn = lin >> 3, k = (lin & 7) * 4;
      float4 f = *(const float4*)(wb + (size_t)nn * E_ + d0 + k);
      alignas(8) bf16 t[4] = {(bf16)f.x, (bf16)f.y, (bf16)f.z, (bf16)f.w};
      *(uint2*)&Bs[nn >> 4][(k >> 4) * 16 + (nn & 15)][k & 15] = *(const uint2*)t;
    }
    __syncthreads();
    v16bf a  = frag_a(&As[wr * 16][0], 40);
    v16bf b0 = frag_b_swz(&Bs[wc * 2 + 0][0][0]);
    v16bf b1 = frag_b_swz(&Bs[wc * 2 + 1][0][0]);
    acc0 = WMMA_BF16(a, b0, acc0);
    acc1 = WMMA_BF16(a, b1, acc1);
  }

  // Epilogue: two 32-row passes staged in LDS, then coalesced b128 stores.
  for (int p = 0; p < 2; ++p) {
    __syncthreads();
    if ((wr >> 1) == p) {
      int rloc = (wr & 1) * 16 + hlf * 8;
#pragma unroll
      for (int v = 0; v < 8; ++v) {
        Fst[rloc + v][wc * 32 + nl]      = acc0[v];
        Fst[rloc + v][wc * 32 + 16 + nl] = acc1[v];
      }
    }
    __syncthreads();
    if (which != 2) {                        // Q,K: row-major [bh][t][e]
      int row = tid >> 3, c8 = tid & 7;
      const float* src = &Fst[row][c8 * 8];
      float4 f0 = *(const float4*)src;
      float4 f1 = *(const float4*)(src + 4);
      alignas(16) bf16 t[8] = {(bf16)f0.x, (bf16)f0.y, (bf16)f0.z, (bf16)f0.w,
                               (bf16)f1.x, (bf16)f1.y, (bf16)f1.z, (bf16)f1.w};
      *(uint4*)(Out + ((size_t)bh * T_ + tbase + p * 32 + row) * E_ + obase + c8 * 8) =
          *(const uint4*)t;
    } else {                                 // V: fragment-order chunks along t
      int o6 = tid >> 2, khlf = (tid >> 1) & 1, kh = tid & 1;
      int og = obase + o6;
      int kb = (tbase + p * 32) >> 5;        // 32-aligned -> one key block
      alignas(16) bf16 t[8];
#pragma unroll
      for (int j = 0; j < 8; ++j)
        t[j] = (bf16)Fst[khlf * 16 + kh * 8 + j][o6];
      size_t base =
          (((size_t)bh * 32 + kb) * 16 + (og >> 4)) * 32 + khlf * 16 + (og & 15);
      *(uint4*)(Vb + base * 16 + kh * 8) = *(const uint4*)t;
    }
  }
}

// ============================================================================
// Kernel 2: causal flash attention per (b,h, 64-row q block).
// Q fragments in registers; K (natural) + V (swizzled) 32-key blocks staged in
// LDS via async global->LDS; online softmax; 8 persistent 16x16 f32
// accumulators per wave; epilogue staged through LDS overlay.
// ============================================================================
__global__ void flash_kernel(const bf16* __restrict__ Qb, const bf16* __restrict__ Kb,
                             const bf16* __restrict__ Vb, bf16* __restrict__ AO) {
  int idx = blockIdx.x;
  int qb = idx & 15, bh = idx >> 4;
  int b = bh >> 3, h = bh & 7;
  int qbase = qb * 64;

  // Manual LDS layout; epilogue f32 staging (33280 B) overlays the loop state.
  __shared__ alignas(16) unsigned char smf[56576];
  bf16 (*Ks)[264]     = (bf16(*)[264])smf;                    // 16896 B
  bf16 (*Vsw)[32][24] = (bf16(*)[32][24])(smf + 16896);       // 24576 B
  float (*Ss)[36]     = (float(*)[36])(smf + 41472);          //  9216 B
  bf16 (*Ps)[40]      = (bf16(*)[40])(smf + 50688);           //  5120 B
  float* row_m        = (float*)(smf + 55808);
  float* row_l        = (float*)(smf + 56064);
  float* row_alpha    = (float*)(smf + 56320);
  float (*Ost)[260]   = (float(*)[260])smf;                   // epilogue 33280 B

  int tid = threadIdx.x, lane = tid & 31, wave = tid >> 5;
  int sr = wave >> 1, sc = wave & 1;   // S-subtile role
  int wc = wave & 1;                   // O role: 128-col half
  int n = lane & 15, hlf = lane >> 4;

  v16bf qfrag[8];
  const bf16* qptr = Qb + ((size_t)bh * T_ + qbase + sr * 16) * E_;
#pragma unroll
  for (int e0 = 0; e0 < 8; ++e0) qfrag[e0] = frag_a(qptr + e0 * 32, E_);

  v8f zero = {};
  v8f acc[8];
#pragma unroll
  for (int s = 0; s < 8; ++s) acc[s] = zero;

  if (tid < 64) { row_m[tid] = -1e30f; row_l[tid] = 0.f; }

  const float scale = 0.0625f;  // 1/sqrt(256)
  int nkb = 2 * qb + 2;         // causal early exit
  for (int kb = 0; kb < nkb; ++kb) {
    int kbase = kb * 32;
    __syncthreads();  // previous iteration's LDS consumers done
    const uint4* ksrc = (const uint4*)(Kb + ((size_t)bh * T_ + kbase) * E_);
    const uint4* vsrc = (const uint4*)(Vb + ((size_t)bh * 32 + kb) * (16 * 32 * 16));
#ifdef HAVE_ASYNC_LDS
#pragma unroll
    for (int i = 0; i < 4; ++i) {
      int lin = i * 256 + tid;
      int r = lin >> 5, c16 = lin & 31;
      __builtin_amdgcn_global_load_async_to_lds_b128(
          (gv4i*)(ksrc + r * 32 + c16), (lv4i*)&Ks[r][c16 * 8], 0, 0);
      int sub = lin >> 6, vl = (lin >> 1) & 31, kk = (lin & 1) * 8;
      __builtin_amdgcn_global_load_async_to_lds_b128(
          (gv4i*)(vsrc + lin), (lv4i*)&Vsw[sub][vl][kk], 0, 0);
    }
#else
#pragma unroll
    for (int i = 0; i < 4; ++i) {
      int lin = i * 256 + tid;
      int r = lin >> 5, c16 = lin & 31;
      *(uint4*)&Ks[r][c16 * 8] = ksrc[r * 32 + c16];
      int sub = lin >> 6, vl = (lin >> 1) & 31, kk = (lin & 1) * 8;
      *(uint4*)&Vsw[sub][vl][kk] = vsrc[lin];
    }
#endif
    if (kb + 1 < nkb) {  // prefetch next K/V block (global_prefetch_b8)
      const char* nk = (const char*)(ksrc + 1024);
      const char* nv = (const char*)(vsrc + 1024);
      __builtin_prefetch(nk + tid * 64, 0, 1);
      __builtin_prefetch(nv + tid * 64, 0, 1);
    }
#ifdef HAVE_ASYNC_LDS
    __builtin_amdgcn_s_wait_asynccnt(0);
#endif
    __syncthreads();

    // S = Q * K^T (this wave's 16x16 subtile)
    v8f s = {};
#pragma unroll
    for (int e0 = 0; e0 < 8; ++e0) {
      v16bf bk = frag_b_t(&Ks[sc * 16][0] + e0 * 32, 264);
      s = WMMA_BF16(qfrag[e0], bk, s);
    }
#pragma unroll
    for (int v = 0; v < 8; ++v)
      Ss[sr * 16 + hlf * 8 + v][sc * 16 + n] = s[v];
    __syncthreads();

    // Online softmax: one thread per q row
    if (tid < 64) {
      int q = qbase + tid;
      float mx = row_m[tid];
      float vals[32];
#pragma unroll
      for (int c4 = 0; c4 < 8; ++c4) {
        float4 vv = *(const float4*)&Ss[tid][c4 * 4];
        vals[c4 * 4 + 0] = vv.x; vals[c4 * 4 + 1] = vv.y;
        vals[c4 * 4 + 2] = vv.z; vals[c4 * 4 + 3] = vv.w;
      }
      float bm = -1e30f;
#pragma unroll
      for (int c = 0; c < 32; ++c) {
        float val = (kbase + c <= q) ? vals[c] * scale : -1e30f;
        vals[c] = val;
        bm = fmaxf(bm, val);
      }
      float mn = fmaxf(mx, bm);
      float alpha = __expf(mx - mn);
      float sum = 0.f;
      alignas(16) bf16 pv[32];
#pragma unroll
      for (int c = 0; c < 32; ++c) {
        float p = __expf(vals[c] - mn);
        sum += p;
        pv[c] = (bf16)p;
      }
#pragma unroll
      for (int c8 = 0; c8 < 4; ++c8)
        *(uint4*)&Ps[tid][c8 * 8] = *(const uint4*)&pv[c8 * 8];
      row_l[tid]     = alpha * row_l[tid] + sum;
      row_m[tid]     = mn;
      row_alpha[tid] = alpha;
    }
    __syncthreads();

    // O = alpha*O + P*V
    float av[8];
#pragma unroll
    for (int v = 0; v < 8; ++v) av[v] = row_alpha[sr * 16 + hlf * 8 + v];
    v16bf pa = frag_a(&Ps[sr * 16][0], 40);
#pragma unroll
    for (int sub = 0; sub < 8; ++sub) {
#pragma unroll
      for (int v = 0; v < 8; ++v) acc[sub][v] = acc[sub][v] * av[v];
      v16bf bv = frag_b_swz(&Vsw[wc * 8 + sub][0][0]);
      acc[sub] = WMMA_BF16(pa, bv, acc[sub]);
    }
  }

  __syncthreads();
  float inv[8];
#pragma unroll
  for (int v = 0; v < 8; ++v) inv[v] = 1.0f / row_l[sr * 16 + hlf * 8 + v];

  // Epilogue: two 32-row passes staged as f32 in LDS (single base address +
  // immediate ds offsets), then coalesced b128 global stores of bf16.
  bf16* aobase = AO + ((size_t)b * T_ + qbase) * HE_ + (size_t)h * E_;
  for (int p = 0; p < 2; ++p) {
    __syncthreads();   // p=0: everyone read row_l; p=1: prev readback done
    if ((sr >> 1) == p) {
      int rloc = (sr & 1) * 16 + hlf * 8;
#pragma unroll
      for (int sub = 0; sub < 8; ++sub) {
        int col = wc * 128 + sub * 16 + n;
#pragma unroll
        for (int v = 0; v < 8; ++v)
          Ost[rloc + v][col] = acc[sub][v] * inv[v];
      }
    }
    __syncthreads();
#pragma unroll
    for (int i = 0; i < 4; ++i) {
      int lin = i * 256 + tid;
      int row = lin >> 5, c8 = lin & 31;
      const float* src = &Ost[row][c8 * 8];
      float4 f0 = *(const float4*)src;
      float4 f1 = *(const float4*)(src + 4);
      alignas(16) bf16 t[8] = {(bf16)f0.x, (bf16)f0.y, (bf16)f0.z, (bf16)f0.w,
                               (bf16)f1.x, (bf16)f1.y, (bf16)f1.z, (bf16)f1.w};
      *(uint4*)(aobase + (size_t)(p * 32 + row) * HE_ + c8 * 8) = *(const uint4*)t;
    }
  }
}

// ============================================================================
// Kernel 3: out = AO (8192x2048, bf16) @ Wu^T (2048x256) + bu   (fp32 out)
// ============================================================================
__global__ void proj_kernel(const bf16* __restrict__ AO, const float* __restrict__ Wu,
                            const float* __restrict__ bu, float* __restrict__ out) {
  int idx = blockIdx.x;
  int nb = idx & 3;       // 256/64 column blocks
  int mb = idx >> 2;      // 8192/64 row blocks
  int rowbase = mb * 64, obase = nb * 64;

  __shared__ alignas(16) bf16 As[64][40];
  __shared__ alignas(16) bf16 Bs[4][32][24];

  int tid = threadIdx.x, lane = tid & 31, wave = tid >> 5;
  int wr = wave >> 1, wc = wave & 1;
  v8f acc0 = {}, acc1 = {};

  for (int k0 = 0; k0 < HE_; k0 += 32) {
    __syncthreads();
    {   // A tile 64x32 bf16, one uint4 per thread
      int r = tid >> 2, c16 = tid & 3;
      *(uint4*)&As[r][c16 * 8] =
          *(const uint4*)(AO + ((size_t)rowbase + r) * HE_ + k0 + c16 * 8);
    }
#pragma unroll
    for (int i = 0; i < 2; ++i) {   // B tile: Wu^T fp32 -> bf16, swizzled
      int lin = i * 256 + tid;
      int nn = lin >> 3, k = (lin & 7) * 4;
      float4 f = *(const float4*)(Wu + ((size_t)obase + nn) * HE_ + k0 + k);
      alignas(8) bf16 t[4] = {(bf16)f.x, (bf16)f.y, (bf16)f.z, (bf16)f.w};
      *(uint2*)&Bs[nn >> 4][(k >> 4) * 16 + (nn & 15)][k & 15] = *(const uint2*)t;
    }
    __syncthreads();
    v16bf a  = frag_a(&As[wr * 16][0], 40);
    v16bf b0 = frag_b_swz(&Bs[wc * 2 + 0][0][0]);
    v16bf b1 = frag_b_swz(&Bs[wc * 2 + 1][0][0]);
    acc0 = WMMA_BF16(a, b0, acc0);
    acc1 = WMMA_BF16(a, b1, acc1);
  }

  int nl = lane & 15, hlf = lane >> 4;
  float* op = out + ((size_t)rowbase + wr * 16 + hlf * 8) * E_ + obase + wc * 32;
#pragma unroll
  for (int v = 0; v < 8; ++v) {
    op[(size_t)v * E_ + nl]      = acc0[v] + bu[obase + wc * 32 + nl];
    op[(size_t)v * E_ + 16 + nl] = acc1[v] + bu[obase + wc * 32 + 16 + nl];
  }
}

extern "C" void kernel_launch(void* const* d_in, const int* in_sizes, int n_in,
                              void* d_out, int out_size, void* d_ws, size_t ws_size,
                              hipStream_t stream) {
  (void)in_sizes; (void)n_in; (void)out_size; (void)ws_size;
  const float* x  = (const float*)d_in[0];
  const float* Wk = (const float*)d_in[1];
  const float* Wq = (const float*)d_in[2];
  const float* Wv = (const float*)d_in[3];
  const float* Wu = (const float*)d_in[4];
  const float* bu = (const float*)d_in[5];
  float* out = (float*)d_out;

  // Workspace: Q,K (natural), V (fragment-order), AO — all bf16, 128 MiB.
  size_t qkv_elems = (size_t)B_ * H_ * T_ * E_;
  bf16* Qb = (bf16*)d_ws;
  bf16* Kb = Qb + qkv_elems;
  bf16* Vb = Kb + qkv_elems;
  bf16* AO = Vb + qkv_elems;

  qkv_kernel<<<dim3(3 * B_ * H_ * (T_ / 64) * (E_ / 64)), 256, 0, stream>>>(
      x, Wq, Wk, Wv, Qb, Kb, Vb);
  flash_kernel<<<dim3(B_ * H_ * (T_ / 64)), 256, 0, stream>>>(Qb, Kb, Vb, AO);
  proj_kernel<<<dim3((B_ * T_ / 64) * (E_ / 64)), 256, 0, stream>>>(AO, Wu, bu, out);
}
